// STQuantize_55490977465092
// MI455X (gfx1250) — matrix operand order repaired and checked by reading, not compiled, not run on hardware
//
#include <hip/hip_runtime.h>
#include <cstdint>

// ---------------------------------------------------------------------------
// VQ-VAE quantization for MI455X (gfx1250).
// B=32, C=256, H=W=32  -> N = 32768 rows;  codes K = 8192, dim 256.
// Core: argmin_k ( ||w_k||^2 - 2 z.w_k ) via bf16x3 split-precision WMMA.
// Weight tiles staged to LDS with CDNA5 async global->LDS loads (double buffer).
// Workspace use: ~42.1 MB.
// d_out layout (floats): [zq 8388608][loss 1][indices 32768][perplexity 1]
// ---------------------------------------------------------------------------

typedef __bf16 bf16;
typedef bf16  v8bf  __attribute__((ext_vector_type(8)));
typedef bf16  v16bf __attribute__((ext_vector_type(16)));
typedef float v8f   __attribute__((ext_vector_type(8)));

#define NROWS   32768
#define NCODES  8192
#define CDIM    256
#define ROWS_PER_BLOCK 128
#define TILE_CODES 16
#define NTILES  (NCODES / TILE_CODES)   // 512

// ---- helpers ---------------------------------------------------------------

// A fragment (16-bit A 16x32, ISA 7.12.2): lane half 0 -> K {0..7,16..23},
// lane half 1 -> K {8..15,24..31}. p points at first 8 elems; second 8 at +16.
__device__ __forceinline__ v16bf load_split16(const bf16* p) {
  v8bf a = *reinterpret_cast<const v8bf*>(p);
  v8bf b = *reinterpret_cast<const v8bf*>(p + 16);
  return __builtin_shufflevector(a, b, 0,1,2,3,4,5,6,7,8,9,10,11,12,13,14,15);
}

// Contiguous 16-element load p[0..15]; unused here (B is loaded with load_split16, A's K order).
__device__ __forceinline__ v16bf load_contig16(const bf16* p) {
  v8bf a = *reinterpret_cast<const v8bf*>(p);
  v8bf b = *reinterpret_cast<const v8bf*>(p + 8);
  return __builtin_shufflevector(a, b, 0,1,2,3,4,5,6,7,8,9,10,11,12,13,14,15);
}

// ---- kernel 1a: transpose z [B,C,H,W] -> zf [N,C], split into bf16 hi/lo ---

__global__ __launch_bounds__(256)
void vq_prep_z_kernel(const float* __restrict__ z,
                      bf16* __restrict__ zf_hi, bf16* __restrict__ zf_lo) {
  __shared__ float tile[64][33];                 // [c][w], padded
  const int b = blockIdx.x >> 5, h = blockIdx.x & 31;
  const int tid = threadIdx.x;
  const int wr = tid & 31, cl = tid >> 5;        // read mapping
  const int w2 = tid >> 3, ci = tid & 7;         // write mapping

  for (int c0 = 0; c0 < CDIM; c0 += 64) {
    #pragma unroll
    for (int cc = cl; cc < 64; cc += 8)
      tile[cc][wr] = z[(((size_t)b * CDIM + c0 + cc) * 32 + h) * 32 + wr];
    __syncthreads();

    const size_t n = ((size_t)(b * 32 + h)) * 32 + w2;
    v8bf hv, lv;
    #pragma unroll
    for (int j = 0; j < 8; ++j) {
      float x = tile[ci * 8 + j][w2];
      bf16 hb = (bf16)x;
      hv[j] = hb;
      lv[j] = (bf16)(x - (float)hb);
    }
    *reinterpret_cast<v8bf*>(zf_hi + n * CDIM + c0 + ci * 8) = hv;
    *reinterpret_cast<v8bf*>(zf_lo + n * CDIM + c0 + ci * 8) = lv;
    __syncthreads();
  }
}

// ---- kernel 1b: split weight into hi/lo, compute ||w_k||^2 -----------------

__global__ __launch_bounds__(256)
void vq_prep_w_kernel(const float* __restrict__ weight,
                      bf16* __restrict__ w_hi, bf16* __restrict__ w_lo,
                      float* __restrict__ wsq) {
  const int k = blockIdx.x, c = threadIdx.x;
  float x = weight[(size_t)k * CDIM + c];
  bf16 hb = (bf16)x;
  w_hi[(size_t)k * CDIM + c] = hb;
  w_lo[(size_t)k * CDIM + c] = (bf16)(x - (float)hb);

  float ss = x * x;
  #pragma unroll
  for (int o = 16; o; o >>= 1) ss += __shfl_down(ss, o, 32);
  __shared__ float part[8];
  if ((c & 31) == 0) part[c >> 5] = ss;
  __syncthreads();
  if (c == 0) {
    float s = 0.f;
    #pragma unroll
    for (int i = 0; i < 8; ++i) s += part[i];
    wsq[k] = s;
  }
}

// ---- kernel 2: WMMA distance sweep + argmin --------------------------------

__device__ __forceinline__ void stage_tile(int t, int buf, int tid,
    const bf16* __restrict__ w_hi, const bf16* __restrict__ w_lo,
    bf16 (*tile)[2][TILE_CODES][CDIM]) {
  const int codeBase = t << 4;
  #pragma unroll
  for (int i = 0; i < 4; ++i) {
    int e    = i * 256 + tid;        // 0..1023  (16-byte units for the tile)
    int term = e >> 9;               // 0 = hi, 1 = lo
    int r    = e & 511;
    int code = r >> 5;
    int kpos = (r & 31) << 3;
    const bf16* src = (term ? w_lo : w_hi) + (size_t)(codeBase + code) * CDIM + kpos;
    unsigned dst = (unsigned)(uintptr_t)(&tile[buf][term][code][kpos]);
    unsigned long long ga = (unsigned long long)(uintptr_t)src;
    asm volatile("global_load_async_to_lds_b128 %0, %1, off"
                 :: "v"(dst), "v"(ga) : "memory");
  }
}

__global__ __launch_bounds__(256)
void vq_argmin_kernel(const bf16* __restrict__ zf_hi, const bf16* __restrict__ zf_lo,
                      const bf16* __restrict__ w_hi, const bf16* __restrict__ w_lo,
                      const float* __restrict__ wsq,
                      int* __restrict__ idx_out, float* __restrict__ idxf_out,
                      int* __restrict__ hist) {
  __shared__ bf16  tile[2][2][TILE_CODES][CDIM]; // 32 KB, double-buffered hi/lo
  __shared__ float red_min[8][8][32];            // 8 KB
  __shared__ int   red_idx[8][8][32];            // 8 KB

  const int tid  = threadIdx.x;
  const int wv   = tid >> 5;
  const int lane = tid & 31;
  const int col  = lane & 15;
  const int half = lane >> 4;
  const int aoff = half << 3;   // A-fragment K offset (0 or 8)
  const int boff = half << 3;   // B-fragment K offset (0 or 8)
  const int rowBase = blockIdx.x * ROWS_PER_BLOCK;

  // Hold this wave's 16-row A fragments (hi+lo) in VGPRs for the whole sweep.
  const int arow = rowBase + (wv << 4) + col;
  v16bf Ah[8], Al[8];
  #pragma unroll
  for (int kc = 0; kc < 8; ++kc) {
    Ah[kc] = load_split16(zf_hi + (size_t)arow * CDIM + kc * 32 + aoff);
    Al[kc] = load_split16(zf_lo + (size_t)arow * CDIM + kc * 32 + aoff);
  }

  float minv[8]; int mini[8];
  #pragma unroll
  for (int r = 0; r < 8; ++r) { minv[r] = 3.4e38f; mini[r] = 0; }

  stage_tile(0, 0, tid, w_hi, w_lo, tile);

  for (int t = 0; t < NTILES; ++t) {
    const int buf = t & 1;
    if (t + 1 < NTILES) {
      stage_tile(t + 1, buf ^ 1, tid, w_hi, w_lo, tile);
      asm volatile("s_wait_asynccnt 4" ::: "memory");  // tile t done (in-order)
    } else {
      asm volatile("s_wait_asynccnt 0" ::: "memory");
    }
    __syncthreads();

    v8f acc = {};
    #pragma unroll
    for (int kc = 0; kc < 8; ++kc) {
      v16bf Bh = load_split16(&tile[buf][0][col][kc * 32 + boff]);
      v16bf Bl = load_split16(&tile[buf][1][col][kc * 32 + boff]);
      // dot ~= hi*hi + lo*hi + hi*lo  (bf16x3 split precision, f32 accumulate)
      acc = __builtin_amdgcn_wmma_f32_16x16x32_bf16(false, Ah[kc], false, Bh,
                                                    (short)0, acc, false, false);
      acc = __builtin_amdgcn_wmma_f32_16x16x32_bf16(false, Al[kc], false, Bh,
                                                    (short)0, acc, false, false);
      acc = __builtin_amdgcn_wmma_f32_16x16x32_bf16(false, Ah[kc], false, Bl,
                                                    (short)0, acc, false, false);
    }

    const int code = (t << 4) + col;
    const float wq = wsq[code];
    #pragma unroll
    for (int r = 0; r < 8; ++r) {
      float s = wq - 2.0f * acc[r];
      bool better = s < minv[r];
      minv[r] = better ? s : minv[r];
      mini[r] = better ? code : mini[r];
    }
    __syncthreads();  // everyone done reading buf before it is restaged
  }

  // Cross-lane argmin: C/D layout -> row m<8: vgpr m, lanes 0-15;
  //                                  row m>=8: vgpr m-8, lanes 16-31.
  #pragma unroll
  for (int r = 0; r < 8; ++r) {
    red_min[wv][r][lane] = minv[r];
    red_idx[wv][r][lane] = mini[r];
  }
  __syncthreads();

  if (tid < 128) {
    const int wvr = tid >> 4;
    const int m   = tid & 15;
    const int r   = m & 7;
    const int lb  = (m >> 3) << 4;
    float best = 3.4e38f; int bi = 0x7fffffff;
    #pragma unroll
    for (int l = 0; l < 16; ++l) {
      float s = red_min[wvr][r][lb + l];
      int   c = red_idx[wvr][r][lb + l];
      if (s < best || (s == best && c < bi)) { best = s; bi = c; }
    }
    const int row = rowBase + (wvr << 4) + m;
    idx_out[row]  = bi;
    idxf_out[row] = (float)bi;
    atomicAdd(&hist[bi], 1);
  }
}

// ---- kernel 3: gather codes -> z_q (NCHW), accumulate SSE ------------------

__global__ __launch_bounds__(256)
void vq_gather_kernel(const float* __restrict__ z, const float* __restrict__ weight,
                      const int* __restrict__ idx, float* __restrict__ zq_out,
                      float* __restrict__ sse_acc) {
  const int b = blockIdx.x >> 5, h = blockIdx.x & 31;
  const int tid = threadIdx.x;
  const int w = tid & 31, cg = tid >> 5;
  const int n = ((b << 5) + h) * 32 + w;
  const int code = idx[n];
  const float* wrow = weight + (size_t)code * CDIM;
  const size_t zb = (((size_t)b * CDIM) * 32 + h) * 32 + w;

  float sse = 0.f;
  #pragma unroll 4
  for (int c = cg; c < CDIM; c += 8) {
    float wv = wrow[c];
    size_t a = zb + (size_t)c * 1024;
    float d = wv - z[a];
    zq_out[a] = wv;
    sse += d * d;
  }
  #pragma unroll
  for (int o = 16; o; o >>= 1) sse += __shfl_down(sse, o, 32);
  __shared__ float part[8];
  if ((tid & 31) == 0) part[tid >> 5] = sse;
  __syncthreads();
  if (tid == 0) {
    float s = 0.f;
    #pragma unroll
    for (int i = 0; i < 8; ++i) s += part[i];
    atomicAdd(sse_acc, s);
  }
}

// ---- kernel 4: loss + perplexity scalars -----------------------------------

__global__ __launch_bounds__(256)
void vq_scalars_kernel(const int* __restrict__ hist, const float* __restrict__ sse,
                       float* __restrict__ loss_out, float* __restrict__ perp_out) {
  const int tid = threadIdx.x;
  float local = 0.f;
  for (int k = tid; k < NCODES; k += 256) {
    float p = (float)hist[k] * (1.0f / (float)NROWS);
    local += p * logf(p + 1e-10f);
  }
  #pragma unroll
  for (int o = 16; o; o >>= 1) local += __shfl_down(local, o, 32);
  __shared__ float part[8];
  if ((tid & 31) == 0) part[tid >> 5] = local;
  __syncthreads();
  if (tid == 0) {
    float s = 0.f;
    #pragma unroll
    for (int i = 0; i < 8; ++i) s += part[i];
    *perp_out = expf(-s);
    // loss = 0.25*mse + mse = 1.25 * SSE / (N*C)   (forward value)
    *loss_out = 1.25f * (*sse) * (1.0f / ((float)NROWS * (float)CDIM));
  }
}

// ---- launcher --------------------------------------------------------------

extern "C" void kernel_launch(void* const* d_in, const int* in_sizes, int n_in,
                              void* d_out, int out_size, void* d_ws, size_t ws_size,
                              hipStream_t stream) {
  (void)in_sizes; (void)n_in; (void)out_size; (void)ws_size;
  const float* z      = (const float*)d_in[0];
  const float* weight = (const float*)d_in[1];

  float* out      = (float*)d_out;
  float* zq_out   = out;                         // 8,388,608
  float* loss_out = out + 8388608;               // 1
  float* idxf_out = out + 8388609;               // 32,768
  float* perp_out = out + 8421377;               // 1

  char* p = (char*)d_ws;
  auto alloc = [&](size_t bytes) {
    char* r = p;
    p += (bytes + 255) & ~(size_t)255;
    return r;
  };
  bf16*  zf_hi = (bf16*)alloc((size_t)NROWS * CDIM * 2);   // 16 MB
  bf16*  zf_lo = (bf16*)alloc((size_t)NROWS * CDIM * 2);   // 16 MB
  bf16*  w_hi  = (bf16*)alloc((size_t)NCODES * CDIM * 2);  // 4 MB
  bf16*  w_lo  = (bf16*)alloc((size_t)NCODES * CDIM * 2);  // 4 MB
  float* wsq   = (float*)alloc(NCODES * 4);
  int*   idxw  = (int*)alloc(NROWS * 4);
  int*   hist  = (int*)alloc(NCODES * 4);
  float* sse   = (float*)alloc(256);

  hipMemsetAsync(hist, 0, NCODES * 4, stream);
  hipMemsetAsync(sse, 0, 4, stream);

  vq_prep_z_kernel<<<1024, 256, 0, stream>>>(z, zf_hi, zf_lo);
  vq_prep_w_kernel<<<NCODES, 256, 0, stream>>>(weight, w_hi, w_lo, wsq);
  vq_argmin_kernel<<<NROWS / ROWS_PER_BLOCK, 256, 0, stream>>>(
      zf_hi, zf_lo, w_hi, w_lo, wsq, idxw, idxf_out, hist);
  vq_gather_kernel<<<1024, 256, 0, stream>>>(z, weight, idxw, zq_out, sse);
  vq_scalars_kernel<<<1, 256, 0, stream>>>(hist, sse, loss_out, perp_out);
}
